// RetNetDecoder_58299886076316
// MI455X (gfx1250) — matrix operand order repaired and checked by reading, not compiled, not run on hardware
//
#include <hip/hip_runtime.h>
#include <math.h>

// ---------------- problem constants (from reference) ----------------
constexpr int   Bb    = 4;
constexpr int   Tt    = 2048;
constexpr int   Dm    = 1024;
constexpr int   Hh    = 4;
constexpr int   VDm   = 2048;
constexpr int   FFNm  = 2048;
constexpr int   Lm    = 6;
constexpr int   Cch   = 512;              // chunk length
constexpr int   KD    = Dm / Hh;          // 256
constexpr int   HD    = VDm / Hh;         // 512
constexpr int   NCH   = Tt / Cch;         // 4 chunks
constexpr int   NBAT  = Bb * NCH * Hh;    // 64 chunk-batches
constexpr int   MROWS = Bb * Tt;          // 8192
constexpr int   VOC1  = 101;
constexpr int   TDIMc = 25;
constexpr int   TPAD  = 2050;             // T + 2
constexpr int   SGRP  = 82;               // TPAD / TDIM
constexpr float EPSc    = 1e-6f;
constexpr float SCALING = 0.0625f;        // KD^-0.5

typedef __bf16 bf16;
typedef __attribute__((ext_vector_type(16))) __bf16 v16bf;
typedef __attribute__((ext_vector_type(8)))  __bf16 v8bf;
typedef __attribute__((ext_vector_type(8)))  float  v8f;

__device__ __forceinline__ float head_decay(int h) {
  // log(1 - 2^(-5-h))
  return log1pf(-exp2f(-5.0f - (float)h));
}
__device__ __forceinline__ int imin(int a, int b) { return a < b ? a : b; }

// =====================================================================
// WMMA GEMM:  Y[M,N] = X[M,K] @ W[N,K]^T   (X,W bf16 packed; Y f32)
// block = 256 thr = 8 waves (4x2); block tile 128x128;
// wave tile 32x64 = 2 A-frags x 4 B-frags = 8 wmma / 12 b128 loads per K step.
// Out-of-range loads are CLAMPED (not predicated) -> EXEC stays all-1s in the
// K loop and no per-iteration zero-fill/exec save-restore is generated; the
// clamped rows/cols only feed output elements that are never stored.
// EPI: 0=store*alpha  1=Y+=  2=exact GELU  3=retention decay-mask
// BOUNDS: predicate stores (only needed for the ragged logits GEMM)
// =====================================================================
template <int EPI, bool BOUNDS>
__global__ void __launch_bounds__(256)
gemm_bf16_kernel(const bf16* __restrict__ Xp, const bf16* __restrict__ Wp,
                 float* __restrict__ Y, int M, int N, int Kd, float alpha,
                 long long bsX, long long bsW, long long bsY, int hmod)
{
  const int bz = blockIdx.z;
  const bf16* Xb = Xp + (long long)bz * bsX;
  const bf16* Wb = Wp + (long long)bz * bsW;
  float*      Yb = Y  + (long long)bz * bsY;

  const int wave = threadIdx.x >> 5;
  const int lane = threadIdx.x & 31;
  const int wm = wave >> 1;             // 0..3
  const int wn = wave & 1;              // 0..1
  const int row0 = blockIdx.y * 128 + wm * 32;
  const int col0 = blockIdx.x * 128 + wn * 64;
  const int l15 = lane & 15;
  const int lh  = lane >> 4;

  v8f zero = {};
  v8f acc[2][4];
#pragma unroll
  for (int i = 0; i < 2; ++i)
#pragma unroll
    for (int j = 0; j < 4; ++j) acc[i][j] = zero;

  // hoisted, clamped fragment base pointers
  const bf16* pa[2];
#pragma unroll
  for (int i = 0; i < 2; ++i) {
    int r = imin(row0 + i * 16 + l15, M - 1);
    pa[i] = Xb + (long long)r * Kd + 8 * lh;
  }
  const bf16* pb[4];
#pragma unroll
  for (int j = 0; j < 4; ++j) {
    int c = imin(col0 + j * 16 + l15, N - 1);
    pb[j] = Wb + (long long)c * Kd + 8 * lh;
  }

  for (int k0 = 0; k0 < Kd; k0 += 32) {
    v16bf a[2], bm[4];
    // A fragment (16x32 bf16): row = lane&15, K runs {8h..8h+7},{16+8h..16+8h+7}
#pragma unroll
    for (int i = 0; i < 2; ++i) {
      v8bf lo = *(const v8bf*)(pa[i] + k0);
      v8bf hi = *(const v8bf*)(pa[i] + k0 + 16);
      v16bf f;
#pragma unroll
      for (int e = 0; e < 8; ++e) { f[e] = lo[e]; f[8 + e] = hi[e]; }
      a[i] = f;
    }
    // B fragment (32x16 bf16): col = lane&15, K runs {8h..8h+7},{16+8h..16+8h+7}
#pragma unroll
    for (int j = 0; j < 4; ++j) {
      v8bf b0 = *(const v8bf*)(pb[j] + k0);
      v8bf b1 = *(const v8bf*)(pb[j] + k0 + 16);
      v16bf f;
#pragma unroll
      for (int e = 0; e < 8; ++e) { f[e] = b0[e]; f[8 + e] = b1[e]; }
      bm[j] = f;
    }
#pragma unroll
    for (int i = 0; i < 2; ++i)
#pragma unroll
      for (int j = 0; j < 4; ++j)
        acc[i][j] = __builtin_amdgcn_wmma_f32_16x16x32_bf16(
            false, a[i], false, bm[j], (short)0, acc[i][j], false, false);
  }

  float decay = 0.f, inv1mr = 0.f;
  if (EPI == 3) {
    int h = bz % hmod;                  // batches ordered (b, chunk, h)
    decay  = head_decay(h);
    inv1mr = 1.0f / (1.0f - expf(decay));
  }

#pragma unroll
  for (int i = 0; i < 2; ++i)
#pragma unroll
    for (int j = 0; j < 4; ++j)
#pragma unroll
      for (int e = 0; e < 8; ++e) {
        int r = row0 + i * 16 + 8 * lh + e;   // C/D layout: row = e + 8h
        int c = col0 + j * 16 + l15;
        if (!BOUNDS || (r < M && c < N)) {
          float v = acc[i][j][e];
          long long off = (long long)r * N + c;
          if (EPI == 0) {
            Yb[off] = v * alpha;
          } else if (EPI == 1) {
            Yb[off] += v;
          } else if (EPI == 2) {
            Yb[off] = 0.5f * v * (1.0f + erff(v * 0.70710678118654752f));
          } else {
            // inner_mask = [i>=j] exp(decay*(i-j)) / sqrt(sum_row mask)
            float m = (r >= c) ? expf(decay * (float)(r - c)) : 0.0f;
            float g = (1.0f - expf(decay * (float)(r + 1))) * inv1mr;
            Yb[off] = v * m * rsqrtf(g);
          }
        }
      }
}

// =====================================================================
// elementwise / packing kernels
// =====================================================================
__global__ void __launch_bounds__(256)
embed_kernel(const int* __restrict__ tok, const float* __restrict__ emb,
             float* __restrict__ X)
{
  long long id = (long long)blockIdx.x * 256 + threadIdx.x;
  if (id >= (long long)MROWS * Dm) return;
  int d = (int)(id % Dm);
  long long r = id / Dm;
  X[id] = 32.0f * emb[(long long)tok[r] * Dm + d];   // sqrt(D)=32
}

__global__ void __launch_bounds__(256)
rmsnorm_kernel(const float* __restrict__ X, const float* __restrict__ W,
               float* __restrict__ Y)
{
  long long row = blockIdx.x;
  const float* x = X + row * Dm;
  float* y = Y + row * Dm;
  float s = 0.f;
  for (int i = threadIdx.x; i < Dm; i += 256) { float v = x[i]; s += v * v; }
  __shared__ float red[256];
  red[threadIdx.x] = s; __syncthreads();
  for (int o = 128; o; o >>= 1) { if (threadIdx.x < o) red[threadIdx.x] += red[threadIdx.x + o]; __syncthreads(); }
  float rn = rsqrtf(red[0] / (float)Dm + EPSc);
  for (int i = threadIdx.x; i < Dm; i += 256) y[i] = x[i] * rn * W[i];
}

__global__ void __launch_bounds__(256)
rmsnorm_pad_kernel(const float* __restrict__ X, const float* __restrict__ W,
                   float* __restrict__ XF)
{
  int row = blockIdx.x;                 // over Bb*TPAD
  int t = row % TPAD, b = row / TPAD;
  float* y = XF + (long long)row * Dm;
  if (t >= Tt) { for (int i = threadIdx.x; i < Dm; i += 256) y[i] = 0.0f; return; }
  const float* x = X + ((long long)b * Tt + t) * Dm;
  float s = 0.f;
  for (int i = threadIdx.x; i < Dm; i += 256) { float v = x[i]; s += v * v; }
  __shared__ float red[256];
  red[threadIdx.x] = s; __syncthreads();
  for (int o = 128; o; o >>= 1) { if (threadIdx.x < o) red[threadIdx.x] += red[threadIdx.x + o]; __syncthreads(); }
  float rn = rsqrtf(red[0] / (float)Dm + EPSc);
  for (int i = threadIdx.x; i < Dm; i += 256) y[i] = x[i] * rn * W[i];
}

__global__ void __launch_bounds__(256)
cvt_f32_bf16_kernel(const float* __restrict__ src, bf16* __restrict__ dst, long long n)
{
  long long id = (long long)blockIdx.x * 256 + threadIdx.x;
  if (id < n) dst[id] = (bf16)src[id];
}

// src: (R x C) f32 row-major -> dst: (C x R) bf16 row-major (i.e. W[K,N] -> Wp[N,K])
__global__ void __launch_bounds__(256)
transpose_pack_kernel(const float* __restrict__ src, bf16* __restrict__ dst,
                      int R, int Ccols)
{
  long long id = (long long)blockIdx.x * 256 + threadIdx.x;
  if (id >= (long long)R * Ccols) return;
  int c = (int)(id % Ccols);
  int r = (int)(id / Ccols);
  dst[(long long)c * R + r] = (bf16)src[id];
}

// rotary theta-shift on q,k; emit chunk-batched bf16 operands
//   QRP/KRP: [(b,n,h)][C][KD],  KRTP: [(b,n,h)][KD][C]
__global__ void __launch_bounds__(256)
theta_shift_kernel(const float* __restrict__ Q, const float* __restrict__ K,
                   bf16* __restrict__ QRP, bf16* __restrict__ KRP,
                   bf16* __restrict__ KRTP)
{
  const int JP = KD / 2;                // 128 angle pairs
  long long id = (long long)blockIdx.x * 256 + threadIdx.x;
  if (id >= (long long)Bb * Tt * Hh * JP) return;
  int j = (int)(id % JP); long long r = id / JP;
  int h = (int)(r % Hh); r /= Hh;
  int t = (int)(r % Tt); int b = (int)(r / Tt);
  // angle = 10000^(-j/(JP-1))
  float ang = expf(-((float)j / (float)(JP - 1)) * 9.210340371976184f);
  float s, c; sincosf((float)t * ang, &s, &c);
  long long qoff = (((long long)b * Tt + t) * Hh + h) * KD + 2 * j;
  float q1 = Q[qoff], q2 = Q[qoff + 1];
  float k1 = K[qoff], k2 = K[qoff + 1];
  float qr1 = q1 * c - q2 * s, qr2 = q2 * c + q1 * s;
  float kr1 = k1 * c - k2 * s, kr2 = k2 * c + k1 * s;
  int n = t / Cch, i = t % Cch;
  long long batch = (long long)(b * NCH + n) * Hh + h;
  long long o = (batch * Cch + i) * KD + 2 * j;
  QRP[o] = (bf16)qr1; QRP[o + 1] = (bf16)qr2;
  KRP[o] = (bf16)kr1; KRP[o + 1] = (bf16)kr2;
  long long ot = (batch * KD + 2 * j) * Cch + i;
  KRTP[ot] = (bf16)kr1; KRTP[ot + Cch] = (bf16)kr2;
}

// v (B,T,VD) -> VCT [(b,n,h)][HD][C] (=vc^T) and VCTW (vc^T * value_inner_decay[c])
__global__ void __launch_bounds__(256)
pack_v_kernel(const float* __restrict__ V, bf16* __restrict__ VCT,
              bf16* __restrict__ VCTW)
{
  long long id = (long long)blockIdx.x * 256 + threadIdx.x;
  if (id >= (long long)NBAT * HD * Cch) return;
  int c = (int)(id % Cch); long long r = id / Cch;
  int d = (int)(r % HD); r /= HD;
  int h = (int)(r % Hh); r /= Hh;
  int n = (int)(r % NCH); int b = (int)(r / NCH);
  float decay = head_decay(h);
  // w[c] = exp(decay*(C-1-c)) * (1-e^decay)/(1-e^(decay*C))
  float w = expf(decay * (float)(Cch - 1 - c)) * (1.0f - expf(decay)) /
            (1.0f - expf(decay * (float)Cch));
  float v = V[((long long)(b * Tt + n * Cch + c)) * VDm + h * HD + d];
  VCT[id]  = (bf16)v;
  VCTW[id] = (bf16)(v * w);
}

// inner_scale[row] = max(sum_j |qk[row,j]|, 1); one wave per row
__global__ void __launch_bounds__(256)
row_abssum_kernel(const float* __restrict__ QK, float* __restrict__ IS)
{
  int wave = threadIdx.x >> 5, lane = threadIdx.x & 31;
  long long row = (long long)blockIdx.x * 8 + wave;
  const float* p = QK + row * Cch;
  float s = 0.f;
  for (int j = lane; j < Cch; j += 32) s += fabsf(p[j]);
  for (int o = 16; o; o >>= 1) s += __shfl_xor(s, o, 32);
  if (lane == 0) IS[row] = fmaxf(s, 1.0f);
}

// per-(b,h) sequential chunk recurrence over kv state
//  emits KVRT [(b,n,h)][HD][KD] = (st/sc)^T in bf16 and CS[(b,n,h)] = sc
__global__ void __launch_bounds__(256)
recurrence_kernel(const float* __restrict__ KV, float* __restrict__ ST,
                  bf16* __restrict__ KVRT, float* __restrict__ CS)
{
  int bh = blockIdx.x;
  int b = bh / Hh, h = bh % Hh;
  float cd = expf(head_decay(h) * (float)Cch);     // cross_decay
  float* st = ST + (long long)bh * KD * HD;
  for (int i = threadIdx.x; i < KD * HD; i += 256) st[i] = 0.0f;
  __syncthreads();
  float sc = 1.0f;
  __shared__ float red[256];
  for (int n = 0; n < NCH; ++n) {
    long long batch = (long long)(b * NCH + n) * Hh + h;
    const float* kv = KV + batch * (long long)KD * HD;
    bf16* out = KVRT + batch * (long long)HD * KD;
    if (threadIdx.x == 0) CS[batch] = sc;
    float colmax = 0.f;
    for (int d = threadIdx.x; d < HD; d += 256) {
      float colsum = 0.f;
      for (int k = 0; k < KD; ++k) {
        float old = st[k * HD + d];
        out[(long long)d * KD + k] = (bf16)(old / sc);
        float nv = old * cd + kv[(long long)k * HD + d];
        st[k * HD + d] = nv;
        colsum += fabsf(nv);
      }
      colmax = fmaxf(colmax, colsum);
    }
    red[threadIdx.x] = colmax;
    __syncthreads();
    for (int o = 128; o; o >>= 1) { if (threadIdx.x < o) red[threadIdx.x] = fmaxf(red[threadIdx.x], red[threadIdx.x + o]); __syncthreads(); }
    sc = fmaxf(red[0], 1.0f);
    __syncthreads();
  }
}

// combine inner/cross with scales + per-(b,t,h) group rms-norm -> RET (B,T,VD)
__global__ void __launch_bounds__(256)
combine_kernel(const float* __restrict__ INNER, const float* __restrict__ CROSS,
               const float* __restrict__ IS, const float* __restrict__ CS,
               float* __restrict__ RET)
{
  int id = blockIdx.x;                  // over Bb*Tt*Hh
  int h = id % Hh; int r = id / Hh;
  int t = r % Tt;  int b = r / Tt;
  int n = t / Cch, i = t % Cch;
  long long batch = (long long)(b * NCH + n) * Hh + h;
  float isc = IS[batch * Cch + i];
  float csc = CS[batch];
  float all = fmaxf(isc, csc);
  float decay = head_decay(h);
  float er = expf(decay);
  float rls = (1.0f - expf(decay * (float)Cch)) / (1.0f - er);        // sum row_last
  float scl = sqrtf((1.0f - expf(decay * (float)(i + 1))) / (1.0f - er));
  float qid = expf(decay * (float)(i + 1)) * rls / scl;               // query_inner_decay
  float cfac = qid * csc / all;
  float ifac = 1.0f / all;
  const float* ip = INNER + (batch * Cch + i) * HD;
  const float* cp = CROSS + (batch * Cch + i) * HD;
  float v0 = ip[threadIdx.x]       * ifac + cp[threadIdx.x]       * cfac;
  float v1 = ip[threadIdx.x + 256] * ifac + cp[threadIdx.x + 256] * cfac;
  __shared__ float red[256];
  red[threadIdx.x] = v0 * v0 + v1 * v1;
  __syncthreads();
  for (int o = 128; o; o >>= 1) { if (threadIdx.x < o) red[threadIdx.x] += red[threadIdx.x + o]; __syncthreads(); }
  float rn = rsqrtf(red[0] / (float)HD + EPSc);
  float* op = RET + ((long long)(b * Tt + t)) * VDm + h * HD;
  op[threadIdx.x]       = v0 * rn;
  op[threadIdx.x + 256] = v1 * rn;
}

__global__ void __launch_bounds__(256)
mul_silu_kernel(float* __restrict__ A, const float* __restrict__ Gt, long long n)
{
  long long id = (long long)blockIdx.x * 256 + threadIdx.x;
  if (id < n) { float g = Gt[id]; A[id] *= g / (1.0f + expf(-g)); }
}

__global__ void __launch_bounds__(256)
mul_kernel(float* __restrict__ A, const float* __restrict__ Bp, long long n)
{
  long long id = (long long)blockIdx.x * 256 + threadIdx.x;
  if (id < n) A[id] *= Bp[id];
}

// gather padded final activations into per-e batches: XOP[e][(b*S+s)][d]
__global__ void __launch_bounds__(256)
pack_out_x_kernel(const float* __restrict__ XF, bf16* __restrict__ XOP)
{
  long long id = (long long)blockIdx.x * 256 + threadIdx.x;
  if (id >= (long long)TDIMc * Bb * SGRP * Dm) return;
  int d = (int)(id % Dm); long long r = id / Dm;
  int rowm = (int)(r % (Bb * SGRP)); int e = (int)(r / (Bb * SGRP));
  int b = rowm / SGRP, s = rowm % SGRP;
  XOP[id] = (bf16)XF[((long long)b * TPAD + s * TDIMc + e) * Dm + d];
}

// scatter YOUT[e][(b*S+s)][v] -> out[b][t][v], t = s*TDIM+e < T
__global__ void __launch_bounds__(256)
scatter_out_kernel(const float* __restrict__ YOUT, float* __restrict__ OUT)
{
  long long id = (long long)blockIdx.x * 256 + threadIdx.x;
  if (id >= (long long)Bb * Tt * VOC1) return;
  int v = (int)(id % VOC1); long long r = id / VOC1;
  int t = (int)(r % Tt); int b = (int)(r / Tt);
  int e = t % TDIMc, s = t / TDIMc;
  OUT[id] = YOUT[((long long)e * (Bb * SGRP) + b * SGRP + s) * VOC1 + v];
}

// =====================================================================
// host
// =====================================================================
static void launch_gemm(int epi, int bounds,
                        const bf16* Xp, const bf16* Wp, float* Y,
                        int M, int N, int K, float alpha,
                        long long bsX, long long bsW, long long bsY,
                        int nbatch, int hmod, hipStream_t stream)
{
  dim3 grid((N + 127) / 128, (M + 127) / 128, nbatch);
  dim3 block(256);
  if (bounds) {
    gemm_bf16_kernel<0, true><<<grid, block, 0, stream>>>(Xp, Wp, Y, M, N, K, alpha, bsX, bsW, bsY, hmod);
    return;
  }
  switch (epi) {
    case 0: gemm_bf16_kernel<0, false><<<grid, block, 0, stream>>>(Xp, Wp, Y, M, N, K, alpha, bsX, bsW, bsY, hmod); break;
    case 1: gemm_bf16_kernel<1, false><<<grid, block, 0, stream>>>(Xp, Wp, Y, M, N, K, alpha, bsX, bsW, bsY, hmod); break;
    case 2: gemm_bf16_kernel<2, false><<<grid, block, 0, stream>>>(Xp, Wp, Y, M, N, K, alpha, bsX, bsW, bsY, hmod); break;
    default: gemm_bf16_kernel<3, false><<<grid, block, 0, stream>>>(Xp, Wp, Y, M, N, K, alpha, bsX, bsW, bsY, hmod); break;
  }
}

static inline long long cdiv(long long a, long long b) { return (a + b - 1) / b; }

extern "C" void kernel_launch(void* const* d_in, const int* in_sizes, int n_in,
                              void* d_out, int out_size, void* d_ws, size_t ws_size,
                              hipStream_t stream) {
  (void)in_sizes; (void)n_in; (void)out_size; (void)ws_size;

  const int*   tokens = (const int*)  d_in[0];
  const float* emb    = (const float*)d_in[1];
  const float* Wq     = (const float*)d_in[2];
  const float* Wk     = (const float*)d_in[3];
  const float* Wv     = (const float*)d_in[4];
  const float* Wg     = (const float*)d_in[5];
  const float* Wo     = (const float*)d_in[6];
  const float* ln1    = (const float*)d_in[7];
  const float* ln2    = (const float*)d_in[8];
  const float* Wgate  = (const float*)d_in[9];
  const float* Wfc1   = (const float*)d_in[10];
  const float* Wfc2   = (const float*)d_in[11];
  const float* lnf    = (const float*)d_in[12];
  const float* Wout   = (const float*)d_in[13];
  float* out = (float*)d_out;

  // ---- workspace layout (256B aligned regions) ----
  char* base = (char*)d_ws;
  size_t off = 0;
  auto alloc = [&](size_t bytes) -> void* {
    void* p = base + off;
    off = (off + bytes + 255) & ~(size_t)255;
    return p;
  };
  float* X    = (float*)alloc((size_t)MROWS * Dm  * 4);
  float* XN   = (float*)alloc((size_t)MROWS * Dm  * 4);
  float* Qb   = (float*)alloc((size_t)MROWS * Dm  * 4);
  float* Kb   = (float*)alloc((size_t)MROWS * Dm  * 4);
  float* Vb   = (float*)alloc((size_t)MROWS * VDm * 4);
  float* Gb   = (float*)alloc((size_t)MROWS * VDm * 4);
  bf16*  XP   = (bf16*) alloc((size_t)MROWS * VDm * 2);
  bf16*  WP   = (bf16*) alloc((size_t)4 * 1024 * 1024 * 2);
  bf16*  QRP  = (bf16*) alloc((size_t)MROWS * Dm  * 2);
  bf16*  KRP  = (bf16*) alloc((size_t)MROWS * Dm  * 2);
  bf16*  KRTP = (bf16*) alloc((size_t)MROWS * Dm  * 2);
  bf16*  VCT  = (bf16*) alloc((size_t)MROWS * VDm * 2);
  bf16*  VCTW = (bf16*) alloc((size_t)MROWS * VDm * 2);
  float* QK   = (float*)alloc((size_t)NBAT * Cch * Cch * 4);
  bf16*  QKB  = (bf16*) alloc((size_t)NBAT * Cch * Cch * 2);
  float* IS   = (float*)alloc((size_t)NBAT * Cch * 4);
  float* INNER= (float*)alloc((size_t)NBAT * Cch * HD * 4);
  float* KV   = (float*)alloc((size_t)NBAT * KD * HD * 4);
  float* ST   = (float*)alloc((size_t)Bb * Hh * KD * HD * 4);
  bf16*  KVRT = (bf16*) alloc((size_t)NBAT * HD * KD * 2);
  float* CS   = (float*)alloc((size_t)NBAT * 4);
  float* CROSS= (float*)alloc((size_t)NBAT * Cch * HD * 4);
  float* RET  = (float*)alloc((size_t)MROWS * VDm * 4);
  float* XF   = (float*)alloc((size_t)Bb * TPAD * Dm * 4);
  bf16*  XOP  = (bf16*) alloc((size_t)TDIMc * Bb * SGRP * Dm * 2);
  float* YOUT = (float*)alloc((size_t)TDIMc * Bb * SGRP * VOC1 * 4);
  float* H1   = QK;     // dead by FFN time
  float* H2   = INNER;  // dead by FFN time

  auto cvt = [&](const float* s, bf16* d, long long n) {
    cvt_f32_bf16_kernel<<<dim3((unsigned)cdiv(n, 256)), dim3(256), 0, stream>>>(s, d, n);
  };
  auto tpack = [&](const float* s, bf16* d, int R, int C) {
    transpose_pack_kernel<<<dim3((unsigned)cdiv((long long)R * C, 256)), dim3(256), 0, stream>>>(s, d, R, C);
  };

  // ---- embedding ----
  embed_kernel<<<dim3((unsigned)cdiv((long long)MROWS * Dm, 256)), dim3(256), 0, stream>>>(tokens, emb, X);

  const long long bCK = (long long)Cch * KD;   // qr/kr batch stride
  const long long bCC = (long long)Cch * Cch;
  const long long bHC = (long long)HD * Cch;
  const long long bCH = (long long)Cch * HD;
  const long long bKH = (long long)KD * HD;
  const long long bHK = (long long)HD * KD;

  for (int l = 0; l < Lm; ++l) {
    const float* wq   = Wq   + (size_t)l * Dm * Dm;
    const float* wk   = Wk   + (size_t)l * Dm * Dm;
    const float* wv   = Wv   + (size_t)l * Dm * VDm;
    const float* wg   = Wg   + (size_t)l * Dm * VDm;
    const float* wo   = Wo   + (size_t)l * VDm * Dm;
    const float* wgt  = Wgate+ (size_t)l * Dm * FFNm;
    const float* wf1  = Wfc1 + (size_t)l * Dm * FFNm;
    const float* wf2  = Wfc2 + (size_t)l * FFNm * Dm;
    const float* L1   = ln1 + (size_t)l * Dm;
    const float* L2   = ln2 + (size_t)l * Dm;

    // ---- retention branch ----
    rmsnorm_kernel<<<dim3(MROWS), dim3(256), 0, stream>>>(X, L1, XN);
    cvt(XN, XP, (long long)MROWS * Dm);
    tpack(wq, WP, Dm, Dm);
    launch_gemm(0, 0, XP, WP, Qb, MROWS, Dm, Dm, 1.0f, 0, 0, 0, 1, 1, stream);
    tpack(wk, WP, Dm, Dm);
    launch_gemm(0, 0, XP, WP, Kb, MROWS, Dm, Dm, SCALING, 0, 0, 0, 1, 1, stream);
    tpack(wv, WP, Dm, VDm);
    launch_gemm(0, 0, XP, WP, Vb, MROWS, VDm, Dm, 1.0f, 0, 0, 0, 1, 1, stream);
    tpack(wg, WP, Dm, VDm);
    launch_gemm(0, 0, XP, WP, Gb, MROWS, VDm, Dm, 1.0f, 0, 0, 0, 1, 1, stream);

    theta_shift_kernel<<<dim3((unsigned)cdiv((long long)Bb * Tt * Hh * (KD / 2), 256)),
                         dim3(256), 0, stream>>>(Qb, Kb, QRP, KRP, KRTP);
    pack_v_kernel<<<dim3((unsigned)cdiv((long long)NBAT * HD * Cch, 256)),
                    dim3(256), 0, stream>>>(Vb, VCT, VCTW);

    // qk = (qr @ kr^T) * inner_mask, batched over (b,chunk,h)
    launch_gemm(3, 0, QRP, KRP, QK, Cch, Cch, KD, 1.0f, bCK, bCK, bCC, NBAT, Hh, stream);
    row_abssum_kernel<<<dim3(NBAT * Cch / 8), dim3(256), 0, stream>>>(QK, IS);
    cvt(QK, QKB, (long long)NBAT * Cch * Cch);
    // inner_raw = qk @ vc
    launch_gemm(0, 0, QKB, VCT, INNER, Cch, HD, Cch, 1.0f, bCC, bHC, bCH, NBAT, 1, stream);
    // kv = kr^T @ (vc * value_inner_decay)
    launch_gemm(0, 0, KRTP, VCTW, KV, KD, HD, Cch, 1.0f, bCK, bHC, bKH, NBAT, 1, stream);
    // sequential cross-chunk state recurrence
    recurrence_kernel<<<dim3(Bb * Hh), dim3(256), 0, stream>>>(KV, ST, KVRT, CS);
    // cross_raw = qr @ kv_rec
    launch_gemm(0, 0, QRP, KVRT, CROSS, Cch, HD, KD, 1.0f, bCK, bHK, bCH, NBAT, 1, stream);
    // combine + group rms-norm
    combine_kernel<<<dim3(Bb * Tt * Hh), dim3(256), 0, stream>>>(INNER, CROSS, IS, CS, RET);
    mul_silu_kernel<<<dim3((unsigned)cdiv((long long)MROWS * VDm, 256)), dim3(256), 0, stream>>>(RET, Gb, (long long)MROWS * VDm);
    cvt(RET, XP, (long long)MROWS * VDm);
    tpack(wo, WP, VDm, Dm);
    launch_gemm(1, 0, XP, WP, X, MROWS, Dm, VDm, 1.0f, 0, 0, 0, 1, 1, stream);  // x += out@Wo

    // ---- FFN branch ----
    rmsnorm_kernel<<<dim3(MROWS), dim3(256), 0, stream>>>(X, L2, XN);
    cvt(XN, XP, (long long)MROWS * Dm);
    tpack(wf1, WP, Dm, FFNm);
    launch_gemm(2, 0, XP, WP, H1, MROWS, FFNm, Dm, 1.0f, 0, 0, 0, 1, 1, stream); // gelu(h@Wfc1)
    tpack(wgt, WP, Dm, FFNm);
    launch_gemm(0, 0, XP, WP, H2, MROWS, FFNm, Dm, 1.0f, 0, 0, 0, 1, 1, stream); // h@Wgate
    mul_kernel<<<dim3((unsigned)cdiv((long long)MROWS * FFNm, 256)), dim3(256), 0, stream>>>(H1, H2, (long long)MROWS * FFNm);
    cvt(H1, XP, (long long)MROWS * FFNm);
    tpack(wf2, WP, FFNm, Dm);
    launch_gemm(1, 0, XP, WP, X, MROWS, Dm, FFNm, 1.0f, 0, 0, 0, 1, 1, stream);  // x += f@Wfc2
  }

  // ---- final norm + output projection ----
  rmsnorm_pad_kernel<<<dim3(Bb * TPAD), dim3(256), 0, stream>>>(X, lnf, XF);
  pack_out_x_kernel<<<dim3((unsigned)cdiv((long long)TDIMc * Bb * SGRP * Dm, 256)),
                      dim3(256), 0, stream>>>(XF, XOP);
  cvt(Wout, WP, (long long)TDIMc * VOC1 * Dm);   // Wout[e] already N x K layout
  launch_gemm(0, 1, XOP, WP, YOUT, Bb * SGRP, VOC1, Dm, 1.0f,
              (long long)(Bb * SGRP) * Dm, (long long)VOC1 * Dm,
              (long long)(Bb * SGRP) * VOC1, TDIMc, 1, stream);
  scatter_out_kernel<<<dim3((unsigned)cdiv((long long)Bb * Tt * VOC1, 256)),
                       dim3(256), 0, stream>>>(YOUT, out);
}
